// GraphAttention_65566970740904
// MI455X (gfx1250) — matrix operand order repaired and checked by reading, not compiled, not run on hardware
//
#include <hip/hip_runtime.h>
#include <stdint.h>

#define B_    4
#define N_    1024
#define FIN_  256
#define FOUT_ 256
#define H_    8
#define HID_  32
#define ALPHA_ 0.2f
#define NEGV_ (-1.0e10f)

typedef __attribute__((ext_vector_type(16))) __bf16 v16bf;
typedef __attribute__((ext_vector_type(8)))  float  v8f;

union Q2V { uint4 q[2]; v16bf v; };
union QB  { uint4 q; __bf16 e[8]; };   // 8 bf16 <-> 16 bytes

static __device__ __forceinline__ __bf16 cvt_bf16(float f) {
  return (__bf16)f;   // native gfx1250 f32->bf16 cvt (RNE)
}

// ---------------------------------------------------------------------------
// K0: one-time W pre-transpose+convert: WT[f][k] = bf16(W[k][f]).
// Coalesced writes (WT index == global thread index).
// ---------------------------------------------------------------------------
__global__ __launch_bounds__(256) void k0_wt(
    const float* __restrict__ W, __bf16* __restrict__ WT)
{
  const int idx = blockIdx.x * 256 + threadIdx.x;   // f*FIN + k
  const int k = idx & (FIN_ - 1);
  const int f = idx >> 8;
  WT[idx] = cvt_bf16(W[(size_t)k * FOUT_ + f]);
}

// ---------------------------------------------------------------------------
// K1: s = h @ W  (per batch: 1024x256 @ 256x256), bf16 WMMA, fp32 accumulate.
// Block = 128 threads (4 waves). Block tile: 64 rows x 64 cols, wave owns 16 rows.
// Writes s (fp32, node-major, for e_i/e_j) and sbT (bf16, [b][h][f][n], packed
// b128 stores; consumed transpose-free by K4).
// ---------------------------------------------------------------------------
__global__ __launch_bounds__(128) void k1_gemm_s(
    const float* __restrict__ h, const __bf16* __restrict__ WT,
    float* __restrict__ s, __bf16* __restrict__ sbT)
{
  __shared__ __align__(16) __bf16 lA[64 * 32];   // [row][k]
  __shared__ __align__(16) __bf16 lB[64 * 32];   // [f][k]
  const int b = blockIdx.z, row0 = blockIdx.y * 64, col0 = blockIdx.x * 64;
  const int t = threadIdx.x, wave = t >> 5, lane = t & 31;
  const int r = lane & 15, half = lane >> 4;

  v8f acc[4] = {};

  for (int k0 = 0; k0 < FIN_; k0 += 32) {
    { // stage A: 64x32 fp32 -> bf16 row-major, packed b128 LDS writes
      const int row = t >> 1, ks = (t & 1) * 16;
      const float* src = h + ((size_t)(b * N_ + row0 + row)) * FIN_ + k0 + ks;
      #pragma unroll
      for (int u = 0; u < 16; u += 8) {
        float4 v0 = *(const float4*)(src + u);
        float4 v1 = *(const float4*)(src + u + 4);
        QB p;
        p.e[0] = cvt_bf16(v0.x); p.e[1] = cvt_bf16(v0.y);
        p.e[2] = cvt_bf16(v0.z); p.e[3] = cvt_bf16(v0.w);
        p.e[4] = cvt_bf16(v1.x); p.e[5] = cvt_bf16(v1.y);
        p.e[6] = cvt_bf16(v1.z); p.e[7] = cvt_bf16(v1.w);
        *(uint4*)(lA + row * 32 + ks + u) = p.q;
      }
    }
    { // stage B: straight copy from pre-transposed WT (b128 in, b128 out)
      const int f = t >> 1, ks = (t & 1) * 16;
      const __bf16* src = WT + (size_t)(col0 + f) * FIN_ + k0 + ks;
      *(uint4*)(lB + f * 32 + ks)     = *(const uint4*)(src);
      *(uint4*)(lB + f * 32 + ks + 8) = *(const uint4*)(src + 8);
    }
    __syncthreads();

    Q2V a;
    const __bf16* arow = lA + (wave * 16 + r) * 32;
    a.q[0] = *(const uint4*)(arow + half * 8);        // K = 8*half .. +7
    a.q[1] = *(const uint4*)(arow + 16 + half * 8);   // K = 16+8*half .. +7
    #pragma unroll
    for (int nt = 0; nt < 4; nt++) {
      Q2V bb;
      const __bf16* brow = lB + (nt * 16 + r) * 32;   // n = r
      bb.q[0] = *(const uint4*)(brow + half * 8);
      bb.q[1] = *(const uint4*)(brow + 16 + half * 8);
      acc[nt] = __builtin_amdgcn_wmma_f32_16x16x32_bf16(
          false, a.v, false, bb.v, (short)0, acc[nt], false, false);
    }
    __syncthreads();
  }

  const int rowb = row0 + wave * 16 + half * 8;       // M = g + 8*half
  #pragma unroll
  for (int nt = 0; nt < 4; nt++) {
    const int col = col0 + nt * 16 + r;               // = hh*HID + f
    const int hh = col >> 5, f = col & 31;
    QB p;
    #pragma unroll
    for (int g = 0; g < 8; g++) {
      const float v = acc[nt][g];
      s[((size_t)(b * N_ + rowb + g)) * FOUT_ + col] = v;
      p.e[g] = cvt_bf16(v);
    }
    // 8 consecutive n values -> one 16B store in [b][h][f][n] layout
    *(uint4*)(sbT + (((size_t)b * H_ + hh) * HID_ + f) * N_ + rowb) = p.q;
  }
}

// ---------------------------------------------------------------------------
// K2: e_i = s . a_left, e_j = s . a_right per (b,n,h).
// e_j stored in both [b,n,h] (for K3) and [b,h,n] (for K4) layouts.
// ---------------------------------------------------------------------------
__global__ __launch_bounds__(256) void k2_edge(
    const float* __restrict__ s, const float* __restrict__ a,
    float* __restrict__ ei, float* __restrict__ ej_bnh, float* __restrict__ ej_bhn)
{
  const int idx = blockIdx.x * blockDim.x + threadIdx.x;  // (b*N + n)
  if (idx >= B_ * N_) return;
  const int b = idx / N_, n = idx % N_;
  const float* srow = s + (size_t)idx * FOUT_;
  #pragma unroll
  for (int hh = 0; hh < H_; hh++) {
    float el = 0.f, er = 0.f;
    #pragma unroll
    for (int f = 0; f < HID_; f++) {
      const float sv = srow[hh * HID_ + f];
      el += sv * a[f];
      er += sv * a[HID_ + f];
    }
    ei[(size_t)idx * H_ + hh]     = el;
    ej_bnh[(size_t)idx * H_ + hh] = er;
    ej_bhn[((size_t)b * H_ + hh) * N_ + n] = er;
  }
}

// ---------------------------------------------------------------------------
// K3: softmax is over axis=1 (the i axis) => per-(b,j,h) column stats.
// Online (m, Z) per thread over i-strides; adj columns read coalesced
// (64 consecutive j per i). Outputs m and 1/Z in [b,h,j] layout.
// ---------------------------------------------------------------------------
__global__ __launch_bounds__(256) void k3_stats(
    const int* __restrict__ adj, const float* __restrict__ ei,
    const float* __restrict__ ej_bnh,
    float* __restrict__ mOut, float* __restrict__ izOut)
{
  __shared__ float sm[4][64][H_];
  __shared__ float sz[4][64][H_];
  const int b = blockIdx.y;
  const int jl = threadIdx.x & 63, ig = threadIdx.x >> 6;
  const int j = blockIdx.x * 64 + jl;

  float ej[H_];
  #pragma unroll
  for (int hh = 0; hh < H_; hh++) ej[hh] = ej_bnh[(size_t)(b * N_ + j) * H_ + hh];

  float m[H_], Z[H_];
  #pragma unroll
  for (int hh = 0; hh < H_; hh++) { m[hh] = -3.0e38f; Z[hh] = 0.f; }

  for (int i = ig; i < N_; i += 4) {
    const int av = adj[((size_t)(b * N_ + i)) * N_ + j];
    const float* eirow = ei + (size_t)(b * N_ + i) * H_;
    #pragma unroll
    for (int hh = 0; hh < H_; hh++) {
      float e = eirow[hh] + ej[hh];
      e = e > 0.f ? e : ALPHA_ * e;
      const float val = av ? e : NEGV_;
      if (val > m[hh]) { Z[hh] = Z[hh] * expf(m[hh] - val) + 1.f; m[hh] = val; }
      else             { Z[hh] += expf(val - m[hh]); }
    }
  }
  #pragma unroll
  for (int hh = 0; hh < H_; hh++) { sm[ig][jl][hh] = m[hh]; sz[ig][jl][hh] = Z[hh]; }
  __syncthreads();
  if (ig == 0) {
    #pragma unroll
    for (int hh = 0; hh < H_; hh++) {
      float mm = m[hh], zz = Z[hh];
      #pragma unroll
      for (int g = 1; g < 4; g++) {
        const float m2 = sm[g][jl][hh], z2 = sz[g][jl][hh];
        if (m2 > mm) { zz = zz * expf(mm - m2) + z2; mm = m2; }
        else         { zz += z2 * expf(m2 - mm); }
      }
      mOut [((size_t)b * H_ + hh) * N_ + j] = mm;
      izOut[((size_t)b * H_ + hh) * N_ + j] = 1.0f / zz;
    }
  }
}

// ---------------------------------------------------------------------------
// K4: h'[b,i,h,:] = sum_j att[b,i,j,h] * s[b,j,h,:]  + bias, via WMMA.
// Block = (b, h, 64 i-rows), 4 waves of 16 rows. K-loop over j in chunks of 32:
// att tile computed on the fly (exp, normalize, pack bf16, b128 LDS writes);
// s tile copied straight from sbT[b][h][f][j] (b128 in/out), shared by all
// 4 waves. 2 WMMAs per chunk.
// ---------------------------------------------------------------------------
__global__ __launch_bounds__(128) void k4_aggregate(
    const int* __restrict__ adj, const __bf16* __restrict__ sbT,
    const float* __restrict__ ei, const float* __restrict__ ej_bhn,
    const float* __restrict__ mIn, const float* __restrict__ izIn,
    const float* __restrict__ bias, float* __restrict__ out)
{
  __shared__ __align__(16) __bf16 lS[32 * 32];        // [f][j]
  __shared__ __align__(16) __bf16 lAtt[4][16 * 32];   // per-wave [i][j]
  const int b = blockIdx.z, hh = blockIdx.y, i0 = blockIdx.x * 64;
  const int t = threadIdx.x, wave = t >> 5, lane = t & 31;
  const int r = lane & 15, half = lane >> 4;
  const int iw = i0 + wave * 16;

  v8f acc0 = {}, acc1 = {};

  const int il = lane >> 1;            // att row this lane computes (0..15)
  const int jb = (lane & 1) * 16;      // att col group (0 or 16)
  const float eiv = ei[(size_t)(b * N_ + iw + il) * H_ + hh];
  const float* ejrow = ej_bhn + ((size_t)b * H_ + hh) * N_;
  const float* mrow  = mIn    + ((size_t)b * H_ + hh) * N_;
  const float* izrow = izIn   + ((size_t)b * H_ + hh) * N_;
  const int*   adjrow = adj + ((size_t)(b * N_ + iw + il)) * N_;

  // staging source row for this thread: fixed f, contiguous in j
  const int fS = t >> 2, jS = (t & 3) * 8;
  const __bf16* srcS = sbT + (((size_t)b * H_ + hh) * HID_ + fS) * N_;

  for (int j0 = 0; j0 < N_; j0 += 32) {
    // stage s tile: one b128 load + one b128 LDS store per thread
    *(uint4*)(lS + fS * 32 + jS) = *(const uint4*)(srcS + j0 + jS);
    { // att tile: exp(leakyrelu(e_i+e_j) masked - m[j]) * invZ[j], packed stores
      const int jj = j0 + jb;
      #pragma unroll
      for (int g = 0; g < 2; g++) {
        QB p;
        #pragma unroll
        for (int k = 0; k < 8; k++) {
          const int j = jj + g * 8 + k;
          const int av = adjrow[j];
          float e = eiv + ejrow[j];
          e = e > 0.f ? e : ALPHA_ * e;
          const float val = av ? e : NEGV_;
          p.e[k] = cvt_bf16(expf(val - mrow[j]) * izrow[j]);
        }
        *(uint4*)(&lAtt[wave][il * 32 + jb + g * 8]) = p.q;
      }
    }
    __syncthreads();

    Q2V a;
    const __bf16* arow = &lAtt[wave][r * 32];
    a.q[0] = *(const uint4*)(arow + half * 8);
    a.q[1] = *(const uint4*)(arow + 16 + half * 8);
    Q2V b0, b1;
    const __bf16* br0 = lS + r * 32;          // n = r, f half 0
    b0.q[0] = *(const uint4*)(br0 + half * 8);
    b0.q[1] = *(const uint4*)(br0 + 16 + half * 8);
    const __bf16* br1 = lS + (16 + r) * 32;   // f half 1
    b1.q[0] = *(const uint4*)(br1 + half * 8);
    b1.q[1] = *(const uint4*)(br1 + 16 + half * 8);

    acc0 = __builtin_amdgcn_wmma_f32_16x16x32_bf16(
        false, a.v, false, b0.v, (short)0, acc0, false, false);
    acc1 = __builtin_amdgcn_wmma_f32_16x16x32_bf16(
        false, a.v, false, b1.v, (short)0, acc1, false, false);
    __syncthreads();
  }

  const float bias0 = bias[hh * HID_ + r];
  const float bias1 = bias[hh * HID_ + 16 + r];
  #pragma unroll
  for (int g = 0; g < 8; g++) {
    const size_t row = (size_t)(b * N_ + iw + half * 8 + g);
    out[row * FOUT_ + hh * HID_ + r]      = acc0[g] + bias0;
    out[row * FOUT_ + hh * HID_ + 16 + r] = acc1[g] + bias1;
  }
}

// ---------------------------------------------------------------------------
extern "C" void kernel_launch(void* const* d_in, const int* in_sizes, int n_in,
                              void* d_out, int out_size, void* d_ws, size_t ws_size,
                              hipStream_t stream) {
  (void)in_sizes; (void)n_in; (void)out_size; (void)ws_size;
  const float* h    = (const float*)d_in[0];
  const int*   adj  = (const int*)  d_in[1];
  const float* W    = (const float*)d_in[2];
  const float* a    = (const float*)d_in[3];
  const float* bias = (const float*)d_in[4];
  float* out = (float*)d_out;

  // ws layout: s fp32 4MB | sbT bf16 2MB | WT bf16 128KB | 5x 128KB fp32
  char* ws = (char*)d_ws;
  float*  s   = (float*) (ws);
  __bf16* sbT = (__bf16*)(ws + (size_t)4 * 1024 * 1024);
  __bf16* WT  = (__bf16*)(ws + (size_t)6 * 1024 * 1024);
  float*  ei  = (float*) (ws + (size_t)6 * 1024 * 1024 + 128 * 1024);
  float*  ejn = ei  + (size_t)B_ * N_ * H_;
  float*  ejh = ejn + (size_t)B_ * N_ * H_;
  float*  mS  = ejh + (size_t)B_ * N_ * H_;
  float*  iz  = mS  + (size_t)B_ * N_ * H_;

  k0_wt       <<<dim3((FOUT_ * FIN_) / 256), 256, 0, stream>>>(W, WT);
  k1_gemm_s   <<<dim3(FOUT_ / 64, N_ / 64, B_), 128, 0, stream>>>(h, WT, s, sbT);
  k2_edge     <<<dim3((B_ * N_) / 256), 256, 0, stream>>>(s, a, ei, ejn, ejh);
  k3_stats    <<<dim3(N_ / 64, B_), 256, 0, stream>>>(adj, ei, ejn, mS, iz);
  k4_aggregate<<<dim3(N_ / 64, H_, B_), 128, 0, stream>>>(adj, sbT, ei, ejh, mS, iz, bias, out);
}
